// UpdateEdgeBlock_20847771255433
// MI455X (gfx1250) — matrix operand order repaired and checked by reading, not compiled, not run on hardware
//
#include <hip/hip_runtime.h>
#include <hip/hip_bf16.h>

#define E_TOT 100000
#define C_CH  128
#define NBAS  16
#define NCPL  11
#define RC_F  5.0f
#define EPS_F 1e-8f
#define PI_F  3.14159265358979f

typedef __attribute__((ext_vector_type(16))) __bf16 v16bf;
typedef __attribute__((ext_vector_type(8)))  float  v8f;

struct alignas(32) Pack32 { uint4 lo, hi; };

// LDS A-matrix layout (bf16, padded row stride)
#define LDA       136     // 272B row stride = 17*16B -> conflict-friendly, 16B aligned
#define ROWS_M0   0       // 16 rows  : m0
#define ROWS_M1   16      // 48 rows  : m1 (row = 16 + a*16 + e)
#define ROWS_M2   64      // 144 rows : m2 (row = 64 + (3a+b)*16 + e)
#define ROWS_MM0  208     // 16 rows  : mixed m0 (input to gate matmuls)
#define N_AROWS   224
#define GSTRIDE   132     // f32 gate row stride
#define WCOLS     (NCPL * C_CH)   // 1408 radial-filter columns

// ws layout: 6 x [128x128] bf16 transposed weights, then Wrt [1408 x 32] bf16
#define WS_WRT_OFF (6 * C_CH * C_CH)
#define WRT_ELEMS  (WCOLS * 32)

// ---- weight prep: transpose+convert U/Wg, and build radial B matrix Wrt ------------
// Wt[f][c]  = (bf16) W[c][f]                       (B fragment wants K contiguous)
// Wrt[col][kb] = kb<16 ? (bf16) W_rad[k][kb][c] : 0   with col = k*128 + c
__global__ void wconv_kernel(const float* __restrict__ U0, const float* __restrict__ U1,
                             const float* __restrict__ U2, const float* __restrict__ G0,
                             const float* __restrict__ G1, const float* __restrict__ G2,
                             const float* __restrict__ W_rad, __bf16* __restrict__ out) {
  int t = blockIdx.x * blockDim.x + threadIdx.x;
  if (t < WS_WRT_OFF) {
    int m = t >> 14;
    int e = t & (C_CH * C_CH - 1);
    int f = e >> 7, c = e & 127;
    const float* src = (m == 0) ? U0 : (m == 1) ? U1 : (m == 2) ? U2
                     : (m == 3) ? G0 : (m == 4) ? G1 : G2;
    out[t] = (__bf16)src[c * C_CH + f];
  } else if (t < WS_WRT_OFF + WRT_ELEMS) {
    int q = t - WS_WRT_OFF;
    int col = q >> 5, kb = q & 31;
    int k = col >> 7, c = col & 127;
    out[t] = (kb < NBAS) ? (__bf16)W_rad[(k * NBAS + kb) * C_CH + c] : (__bf16)0.0f;
  }
}

// ---------------- WMMA fragment helpers (layouts per CDNA5 ISA §7.12.2) ----
__device__ __forceinline__ v16bf load_fragA(const __bf16* base, int rowBase, int lda,
                                            int kbase, int lane) {
  // 16-bit A 16x32: lanes 0-15 -> K = kb+0..7 (lo), kb+16..23 (hi); lanes 16-31 shifted by 8
  int m = lane & 15, hi = lane >> 4;
  const __bf16* r = base + (rowBase + m) * lda;
  Pack32 p;
  p.lo = *(const uint4*)(r + kbase + hi * 8);
  p.hi = *(const uint4*)(r + kbase + 16 + hi * 8);
  return __builtin_bit_cast(v16bf, p);
}

__device__ __forceinline__ v16bf load_fragB(const __bf16* Wt, int ntile, int ldb,
                                            int kbase, int lane) {
  // 16-bit B 32x16: lane%16 = column N; lanes 0-15 -> K = kb+0..7, kb+16..23; lanes 16-31 shifted by 8
  int n = lane & 15, hi = lane >> 4;
  const __bf16* r = Wt + (ntile * 16 + n) * ldb + kbase + hi * 8;
  Pack32 p;
  p.lo = *(const uint4*)(r);
  p.hi = *(const uint4*)(r + 16);
  return __builtin_bit_cast(v16bf, p);
}

__device__ __forceinline__ v8f wmma_tile(const __bf16* smA, int rowBase,
                                         const __bf16* Wt, int ntile, int lane, v8f acc) {
  #pragma unroll
  for (int ks = 0; ks < 4; ++ks) {   // K = 128 in 4 steps of 32
    v16bf a = load_fragA(smA, rowBase, LDA, ks * 32, lane);
    v16bf b = load_fragB(Wt, ntile, C_CH, ks * 32, lane);
    acc = __builtin_amdgcn_wmma_f32_16x16x32_bf16(false, a, false, b, (short)0, acc,
                                                  false, false);
  }
  return acc;
}

// ---------------- fused edge-update kernel: 16 edges per workgroup, 8 waves ----------
__global__ __launch_bounds__(256)
void edge_update_kernel(const float* __restrict__ node0, const float* __restrict__ node1,
                        const float* __restrict__ node2, const float* __restrict__ edge0,
                        const float* __restrict__ edge1, const float* __restrict__ edge2,
                        const float* __restrict__ rij,
                        const float* __restrict__ b0,    const float* __restrict__ bg0,
                        const float* __restrict__ bg1,   const float* __restrict__ bg2,
                        const int* __restrict__ idx_j,   const __bf16* __restrict__ wts,
                        float* __restrict__ out0, float* __restrict__ out1,
                        float* __restrict__ out2) {
  extern __shared__ char smem[];
  __bf16* smA    = (__bf16*)smem;                              // 224*136 bf16 (60928 B)
  __bf16* smW    = (__bf16*)(smem + N_AROWS * LDA * 2);        // 16*1408 bf16 (45056 B)
  __bf16* smRbfA = smW + 16 * WCOLS;                           // 16*32 bf16 (1024 B)
  float*  smG    = (float*)(smRbfA + 16 * 32);                 // 3*16*132 f32 (25344 B)
  float*  smRhat = smG + 3 * 16 * GSTRIDE;                     // 16*4 f32
  int*    smIdx  = (int*)(smRhat + 16 * 4);                    // 16

  const int e0    = blockIdx.x * 16;
  const int tid   = threadIdx.x;
  const int lane  = tid & 31;
  const int wave  = tid >> 5;
  const int ncol  = lane & 15;
  const int mhalf = lane >> 4;

  // ---- stage 0: geometry + radial basis, staged directly as a 16x32 bf16 A tile ----
  {
    int e = tid >> 4, b = tid & 15;
    int eg = e0 + e;
    float rx = rij[eg * 3 + 0], ry = rij[eg * 3 + 1], rz = rij[eg * 3 + 2];
    float d = sqrtf(rx * rx + ry * ry + rz * rz);
    float inv = 1.0f / (d + EPS_F);
    if (b < 3) smRhat[e * 4 + b] = (b == 0 ? rx : b == 1 ? ry : rz) * inv;
    if (b == 3) smIdx[e] = idx_j[eg];
    float env = (d < RC_F) ? 0.5f * (__cosf(PI_F * d / RC_F) + 1.0f) : 0.0f;
    float basis = __sinf((float)(b + 1) * PI_F * d / RC_F) * inv * env;
    smRbfA[e * 32 + b]      = (__bf16)basis;   // K = 0..15 : rbf
    smRbfA[e * 32 + 16 + b] = (__bf16)0.0f;    // K = 16..31: zero pad
  }
  __syncthreads();

  // ---- stage 0.5: radial filters as WMMA: w[16 x 1408] = rbf[16x16] @ W_rad[16x1408] ----
  {
    const __bf16* Wrt = wts + WS_WRT_OFF;            // [1408 x 32] bf16, K-contiguous
    v16bf a = load_fragA(smRbfA, 0, 32, 0, lane);    // loop-invariant A fragment
    #pragma unroll 1
    for (int i = 0; i < 11; ++i) {                   // 88 col tiles / 8 waves
      int ntile = wave * 11 + i;
      v16bf bfr = load_fragB(Wrt, ntile, 32, 0, lane);
      v8f acc = { 0.f, 0.f, 0.f, 0.f, 0.f, 0.f, 0.f, 0.f };
      acc = __builtin_amdgcn_wmma_f32_16x16x32_bf16(false, a, false, bfr, (short)0, acc,
                                                    false, false);
      #pragma unroll
      for (int v = 0; v < 8; ++v) {
        int M = v + mhalf * 8;                       // M = edge, N = filter column
        smW[M * WCOLS + ntile * 16 + ncol] = (__bf16)acc[v];
      }
    }
  }
  __syncthreads();

  // ---- stage 1: build message tensors m0/m1/m2 as bf16 A-matrix rows in LDS ----
  {
    int c = tid & 127;                 // channel is constant per thread
    #pragma unroll 1
    for (int i = 0; i < 8; ++i) {      // 2048 (e,c) pairs / 256 threads
      int p = tid + 256 * i;
      int e = p >> 7;
      int j = smIdx[e];
      float rh0 = smRhat[e * 4 + 0], rh1 = smRhat[e * 4 + 1], rh2 = smRhat[e * 4 + 2];
      float w[NCPL];
      #pragma unroll
      for (int k = 0; k < NCPL; ++k) w[k] = (float)smW[e * WCOLS + k * C_CH + c];
      float x0 = node0[j * C_CH + c];
      float x1[3], x2[9];
      #pragma unroll
      for (int a = 0; a < 3; ++a) x1[a] = node1[(j * C_CH + c) * 3 + a];
      #pragma unroll
      for (int a = 0; a < 9; ++a) x2[a] = node2[(j * C_CH + c) * 9 + a];
      float rh[3] = { rh0, rh1, rh2 };
      float dot1 = x1[0] * rh0 + x1[1] * rh1 + x1[2] * rh2;          // x1 . rhat
      float x2r[3];
      #pragma unroll
      for (int a = 0; a < 3; ++a)
        x2r[a] = x2[a * 3 + 0] * rh0 + x2[a * 3 + 1] * rh1 + x2[a * 3 + 2] * rh2;  // x2 @ rhat
      float quad = x2r[0] * rh0 + x2r[1] * rh1 + x2r[2] * rh2;       // x2 : R2

      float m0 = w[0] * x0 + w[4] * dot1 + w[9] * quad;
      smA[(ROWS_M0 + e) * LDA + c] = (__bf16)m0;
      #pragma unroll
      for (int a = 0; a < 3; ++a) {
        float m1 = w[1] * x0 * rh[a] + w[3] * x1[a] + w[6] * dot1 * rh[a] + w[8] * x2r[a];
        smA[(ROWS_M1 + a * 16 + e) * LDA + c] = (__bf16)m1;
      }
      #pragma unroll
      for (int a = 0; a < 3; ++a) {
        #pragma unroll
        for (int bb = 0; bb < 3; ++bb) {
          float m2 = w[2] * x0 * rh[a] * rh[bb] + w[5] * x1[a] * rh[bb]
                   + w[7] * x2[a * 3 + bb]      + w[10] * x2r[a] * rh[bb];
          smA[(ROWS_M2 + (a * 3 + bb) * 16 + e) * LDA + c] = (__bf16)m2;
        }
      }
    }
  }
  __syncthreads();

  // ---- stage 2: mm0 = m0 @ U0 + b0  (8 col tiles, one per wave) ----
  {
    int ntile = wave;
    float bias = b0[ntile * 16 + ncol];
    v8f acc = { bias, bias, bias, bias, bias, bias, bias, bias };
    acc = wmma_tile(smA, ROWS_M0, wts /* U0t */, ntile, lane, acc);
    #pragma unroll
    for (int v = 0; v < 8; ++v) {
      int M = v + mhalf * 8;           // C/D layout: M = vgpr + 8*(lane>=16), N = lane%16
      smA[(ROWS_MM0 + M) * LDA + ntile * 16 + ncol] = (__bf16)acc[v];
    }
  }
  __syncthreads();

  // ---- stage 3a: gates silu(mm0 @ Wg{0,1,2} + bg); out0 written for g0 ----
  {
    #pragma unroll 1
    for (int i = 0; i < 3; ++i) {      // 24 gate tiles / 8 waves
      int t = wave * 3 + i;
      int g = t >> 3, ntile = t & 7;
      const __bf16* Wt = wts + (3 + g) * C_CH * C_CH;
      const float* bg = (g == 0) ? bg0 : (g == 1) ? bg1 : bg2;
      float bias = bg[ntile * 16 + ncol];
      v8f acc = { bias, bias, bias, bias, bias, bias, bias, bias };
      acc = wmma_tile(smA, ROWS_MM0, Wt, ntile, lane, acc);
      #pragma unroll
      for (int v = 0; v < 8; ++v) {
        int M = v + mhalf * 8;
        float x = acc[v];
        float s = x / (1.0f + __expf(-x));
        smG[(g * 16 + M) * GSTRIDE + ntile * 16 + ncol] = s;
        if (g == 0) {
          int idx = (e0 + M) * C_CH + ntile * 16 + ncol;
          out0[idx] = edge0[idx] + s;
        }
      }
    }
  }
  __syncthreads();

  // ---- stage 3b: n1 = m1@U1, n2 = m2@U2; fuse gate*(.) + edge residual + store ----
  {
    #pragma unroll 1
    for (int i = 0; i < 12; ++i) {     // (24 + 72) tiles / 8 waves; uniform per wave
      int t = wave * 12 + i;
      bool is2 = t >= 24;
      int tt = is2 ? t - 24 : t;
      int comp = tt >> 3, ntile = tt & 7;
      int rowBase = is2 ? (ROWS_M2 + comp * 16) : (ROWS_M1 + comp * 16);
      const __bf16* Wt = wts + (is2 ? 2 : 1) * C_CH * C_CH;
      v8f acc = { 0.f, 0.f, 0.f, 0.f, 0.f, 0.f, 0.f, 0.f };
      acc = wmma_tile(smA, rowBase, Wt, ntile, lane, acc);
      int g = is2 ? 2 : 1;
      int nc = is2 ? 9 : 3;
      const float* ein = is2 ? edge2 : edge1;
      float* eout = is2 ? out2 : out1;
      #pragma unroll
      for (int v = 0; v < 8; ++v) {
        int M = v + mhalf * 8;
        int f = ntile * 16 + ncol;
        float gate = smG[(g * 16 + M) * GSTRIDE + f];
        int idx = ((e0 + M) * C_CH + f) * nc + comp;
        eout[idx] = ein[idx] + acc[v] * gate;
      }
    }
  }
}

extern "C" void kernel_launch(void* const* d_in, const int* in_sizes, int n_in,
                              void* d_out, int out_size, void* d_ws, size_t ws_size,
                              hipStream_t stream) {
  const float* node0 = (const float*)d_in[0];
  const float* node1 = (const float*)d_in[1];
  const float* node2 = (const float*)d_in[2];
  const float* edge0 = (const float*)d_in[3];
  const float* edge1 = (const float*)d_in[4];
  const float* edge2 = (const float*)d_in[5];
  const float* rij   = (const float*)d_in[6];
  const float* W_rad = (const float*)d_in[7];
  const float* U0    = (const float*)d_in[8];
  const float* b0    = (const float*)d_in[9];
  const float* U1    = (const float*)d_in[10];
  const float* U2    = (const float*)d_in[11];
  const float* Wg0   = (const float*)d_in[12];
  const float* bg0   = (const float*)d_in[13];
  const float* Wg1   = (const float*)d_in[14];
  const float* bg1   = (const float*)d_in[15];
  const float* Wg2   = (const float*)d_in[16];
  const float* bg2   = (const float*)d_in[17];
  const int*   idx_j = (const int*)d_in[18];

  __bf16* wts = (__bf16*)d_ws;   // 6x16K bf16 weights + 1408x32 bf16 Wrt (~280 KB)
  float* out0 = (float*)d_out;
  float* out1 = out0 + (size_t)E_TOT * C_CH;
  float* out2 = out0 + (size_t)E_TOT * C_CH * 4;

  int wtot = WS_WRT_OFF + WRT_ELEMS;
  wconv_kernel<<<(wtot + 255) / 256, 256, 0, stream>>>(
      U0, U1, U2, Wg0, Wg1, Wg2, W_rad, wts);

  size_t smbytes = (size_t)N_AROWS * LDA * 2 + (size_t)(16 * WCOLS + 16 * 32) * 2 +
                   (size_t)(3 * 16 * GSTRIDE + 16 * 4 + 16) * 4;  // ~129.6 KB
  edge_update_kernel<<<E_TOT / 16, 256, smbytes, stream>>>(
      node0, node1, node2, edge0, edge1, edge2, rij,
      b0, bg0, bg1, bg2, idx_j, wts, out0, out1, out2);
}
